// AttractorNetwork_10471130267720
// MI455X (gfx1250) — matrix operand order repaired and checked by reading, not compiled, not run on hardware
//
#include <hip/hip_runtime.h>

// CDNA5 / gfx1250, wave32.
typedef __attribute__((ext_vector_type(16))) _Float16 v16h;
typedef __attribute__((ext_vector_type(8)))  float    v8f;

#define N_NEURONS   2048
#define BATCH       128
#define STEPS       200
#define CUE_DUR     10
#define HALF_STEPS  100
#define DECAY       0.95122942450071400909f   // exp(-1/20)
#define INHIB_PER_N (0.1f / 2048.0f)

// ---------------------------------------------------------------------------
// Prologue A: build split-precision transposed effective weights.
// w_eff[k][n] = (W[k][n] - inh/n) * (k != n)
// Stored column-major (wT[n][k]) so B-matrix WMMA fragments are contiguous.
// w_hi = f16(w); w_lo = f16(w - w_hi). Since spikes are exactly 0/1, the
// two-term f16 WMMA reproduces the f32 GEMM to ~2^-22 relative error.
// ---------------------------------------------------------------------------
__global__ __launch_bounds__(256)
void build_weights_kernel(const float* __restrict__ W,
                          _Float16* __restrict__ whiT,
                          _Float16* __restrict__ wloT) {
    int gid = blockIdx.x * blockDim.x + threadIdx.x;   // linear over n*2048 + k
    int n = gid >> 11;        // output neuron (column of w_eff)
    int k = gid & 2047;       // presynaptic neuron (row of w_eff)
    float w = (k == n) ? 0.0f : (W[k * N_NEURONS + n] - INHIB_PER_N);
    _Float16 hi = (_Float16)w;
    whiT[gid] = hi;
    wloT[gid] = (_Float16)(w - (float)hi);
}

// ---------------------------------------------------------------------------
// Prologue B: zero state. (d_out is poisoned by the harness; we own it.)
// ---------------------------------------------------------------------------
__global__ __launch_bounds__(256)
void init_state_kernel(float* __restrict__ mem,
                       _Float16* __restrict__ spk0,
                       _Float16* __restrict__ spk1,
                       float* __restrict__ out) {
    int gid = blockIdx.x * blockDim.x + threadIdx.x;
    mem[gid]  = 0.0f;
    spk0[gid] = (_Float16)0.0f;
    spk1[gid] = (_Float16)0.0f;
    out[gid]  = 0.0f;
}

// ---------------------------------------------------------------------------
// One simulation step: rec = spkA @ w_eff  (WMMA), then fused LIF update.
// Grid: 64 blocks x 256 threads (8 waves).
// Wave w owns M-tiles {2*(w&3), 2*(w&3)+1} and N-tile 2*blockIdx + (w>>2):
// each B (w_hi/w_lo) fragment load feeds TWO WMMAs -> 2 b128 loads per WMMA,
// and the c0/c1 chains interleave to hide WMMA->WMMA RAW latency.
// ---------------------------------------------------------------------------
__global__ __launch_bounds__(256)
void step_kernel(const _Float16* __restrict__ spkA,   // [128][2048] prev spikes
                 _Float16* __restrict__ spkB,         // [128][2048] next spikes
                 const _Float16* __restrict__ whiT,   // [2048 n][2048 k]
                 const _Float16* __restrict__ wloT,   // [2048 n][2048 k]
                 const float* __restrict__ cue,       // [128][2048]
                 const float* __restrict__ noise_t,   // noise + t*B*N
                 float* __restrict__ mem,             // [128][2048]
                 float* __restrict__ out,             // [128][2048] activity acc
                 int t) {
    const int wave  = threadIdx.x >> 5;
    const int lane  = threadIdx.x & 31;
    const int hi16  = lane >> 4;              // which 16-lane half
    const int lrow  = lane & 15;
    const int mT0   = 2 * (wave & 3);         // first of two M tiles
    const int nTile = blockIdx.x * 2 + (wave >> 2);   // 0..127

    // A fragment source: row m = mTile*16 + lrow; per-lane K chunks
    // {kbase..kbase+7, kbase+16..kbase+23} with kbase = k0 + 8*hi16.
    const _Float16* arow0 = spkA + (size_t)(mT0 * 16 + lrow) * N_NEURONS;
    const _Float16* arow1 = arow0 + (size_t)16 * N_NEURONS;
    const int aoff = hi16 * 8;
    // B fragment source: column n = nTile*16 + lrow; K chunks
    // {kbase..kbase+7, kbase+16..kbase+23} with kbase = k0 + 8*hi16.
    const size_t bbase = (size_t)(nTile * 16 + lrow) * N_NEURONS + hi16 * 8;
    const _Float16* bhi = whiT + bbase;
    const _Float16* blo = wloT + bbase;

    union Frag { v16h v; float4 q[2]; };

    v8f c0 = {};
    v8f c1 = {};
    #pragma unroll 2
    for (int k0 = 0; k0 < N_NEURONS; k0 += 32) {
        Frag a0, a1, bh, bl;
        a0.q[0] = *(const float4*)(arow0 + k0 + aoff);
        a0.q[1] = *(const float4*)(arow0 + k0 + aoff + 16);
        a1.q[0] = *(const float4*)(arow1 + k0 + aoff);
        a1.q[1] = *(const float4*)(arow1 + k0 + aoff + 16);
        bh.q[0] = *(const float4*)(bhi + k0);
        bh.q[1] = *(const float4*)(bhi + k0 + 16);
        bl.q[0] = *(const float4*)(blo + k0);
        bl.q[1] = *(const float4*)(blo + k0 + 16);
        // D = A*B_lo + (A*B_hi + C): exact 0/1 A makes this ~f32-accurate.
        c0 = __builtin_amdgcn_wmma_f32_16x16x32_f16(
                 false, a0.v, false, bh.v, (short)0, c0, false, false);
        c1 = __builtin_amdgcn_wmma_f32_16x16x32_f16(
                 false, a1.v, false, bh.v, (short)0, c1, false, false);
        c0 = __builtin_amdgcn_wmma_f32_16x16x32_f16(
                 false, a0.v, false, bl.v, (short)0, c0, false, false);
        c1 = __builtin_amdgcn_wmma_f32_16x16x32_f16(
                 false, a1.v, false, bl.v, (short)0, c1, false, false);
    }

    // Prefetch next step's noise slice for rows this wave touches.
    if (t + 1 < STEPS) {
        __builtin_prefetch(noise_t + BATCH * N_NEURONS
                           + (size_t)(mT0 * 16 + lrow) * N_NEURONS
                           + nTile * 16, 0, 1);
    }

    // Fused LIF epilogue straight out of the C fragments:
    // VGPR r, lanes 0-15 -> M = r ; lanes 16-31 -> M = r + 8 ; N = lrow.
    const bool cueOn = (t < CUE_DUR);
    const bool accOn = (t >= HALF_STEPS);
    const int  col   = nTile * 16 + lrow;
    #pragma unroll
    for (int tile = 0; tile < 2; ++tile) {
        const v8f& c = tile ? c1 : c0;
        const int mBase = (mT0 + tile) * 16 + hi16 * 8;
        #pragma unroll
        for (int r = 0; r < 8; ++r) {
            const size_t idx = (size_t)(mBase + r) * N_NEURONS + col;
            // Noise is streamed exactly once: non-temporal, keep weights in L2.
            float nz = __builtin_nontemporal_load(noise_t + idx);
            float mm = mem[idx] * DECAY + c[r]
                     + (cueOn ? cue[idx] : 0.0f) + nz;
            float s  = (mm >= 1.0f) ? 1.0f : 0.0f;   // spike + hard reset
            mem[idx] = mm * (1.0f - s);
            if (accOn) out[idx] += s;
            spkB[idx] = (_Float16)s;
        }
    }
}

// ---------------------------------------------------------------------------
// Epilogue: mean activity = acc / (steps/2)
// ---------------------------------------------------------------------------
__global__ __launch_bounds__(256)
void scale_out_kernel(float* __restrict__ out) {
    int gid = blockIdx.x * blockDim.x + threadIdx.x;
    out[gid] *= (1.0f / (float)HALF_STEPS);
}

// ---------------------------------------------------------------------------
extern "C" void kernel_launch(void* const* d_in, const int* in_sizes, int n_in,
                              void* d_out, int out_size, void* d_ws, size_t ws_size,
                              hipStream_t stream) {
    const float* cue   = (const float*)d_in[0];   // [128, 2048]
    const float* W     = (const float*)d_in[1];   // [2048, 2048]
    const float* noise = (const float*)d_in[2];   // [200, 128, 2048]
    // d_in[3] = steps (200), d_in[4] = cue_duration (10): fixed by reference.

    char* ws = (char*)d_ws;
    _Float16* whiT = (_Float16*)(ws);                                   // 8 MB
    _Float16* wloT = (_Float16*)(ws + (size_t)8  * 1024 * 1024);        // 8 MB
    _Float16* spk0 = (_Float16*)(ws + (size_t)16 * 1024 * 1024);        // 512 KB
    _Float16* spk1 = (_Float16*)(ws + (size_t)16 * 1024 * 1024 + 512 * 1024);
    float*    mem  = (float*)   (ws + (size_t)17 * 1024 * 1024);        // 1 MB
    float*    out  = (float*)d_out;

    build_weights_kernel<<<(N_NEURONS * N_NEURONS) / 256, 256, 0, stream>>>(
        W, whiT, wloT);
    init_state_kernel<<<(BATCH * N_NEURONS) / 256, 256, 0, stream>>>(
        mem, spk0, spk1, out);

    _Float16* bufs[2] = { spk0, spk1 };
    for (int t = 0; t < STEPS; ++t) {
        step_kernel<<<64, 256, 0, stream>>>(
            bufs[t & 1], bufs[(t + 1) & 1], whiT, wloT,
            cue, noise + (size_t)t * BATCH * N_NEURONS, mem, out, t);
    }

    scale_out_kernel<<<(BATCH * N_NEURONS) / 256, 256, 0, stream>>>(out);
}
